// FixLSTM_27324581937437
// MI455X (gfx1250) — matrix operand order repaired and checked by reading, not compiled, not run on hardware
//
#include <hip/hip_runtime.h>

// ---------------------------------------------------------------------------
// 2-layer LSTM for MI455X (gfx1250), wave32 + v_wmma_f32_16x16x32_f16.
//   B=32, S=1024, H=512, 4H=2048, num_layers=2 (per reference setup_inputs).
// Phase 1 (per layer): G[s,b,:] = X @ Wx   (big parallel WMMA GEMM, f16 in / f32 acc)
// Phase 2 (per layer): persistent cooperative kernel walks t=0..S-1,
//   gates = G[t] + h@U (WMMA, C seeded from G), LSTM cell elementwise,
//   h double-buffered f16 in global (L2-resident), c persistent in LDS.
// ---------------------------------------------------------------------------

typedef __attribute__((ext_vector_type(16))) _Float16 v16h;
typedef __attribute__((ext_vector_type(8)))  _Float16 v8h;
typedef __attribute__((ext_vector_type(8)))  float    v8f;

#define B_    32
#define S_    1024
#define H_    512
#define G4H_  2048
#define M_    (B_ * S_)      // 32768
#define NWG_REC 32

__device__ __forceinline__ float sigmoidf_(float x) {
    return 1.0f / (1.0f + __expf(-x));
}

// Grid-wide barrier for the persistent recurrence kernel (NWG_REC blocks).
__device__ __forceinline__ void gridbar(unsigned* cnt, unsigned idx) {
    __threadfence();
    __syncthreads();
    if (threadIdx.x == 0) {
        atomicAdd(cnt, 1u);
        const unsigned target = idx * NWG_REC;
        while (__hip_atomic_load(cnt, __ATOMIC_RELAXED, __HIP_MEMORY_SCOPE_AGENT) < target) {
            __builtin_amdgcn_s_sleep(2);
        }
    }
    __syncthreads();
    __threadfence();
}

// ---- prep kernels ---------------------------------------------------------

__global__ __launch_bounds__(64)
void k_zero_counters(unsigned* c) {
    if (threadIdx.x < 2) c[threadIdx.x] = 0u;
}

__global__ __launch_bounds__(256)
void k_f32_to_f16(const float* __restrict__ in, _Float16* __restrict__ out, int n) {
    int i = blockIdx.x * 256 + threadIdx.x;
    if (i < n) out[i] = (_Float16)in[i];
}

// in: [512, 2048] fp32 row-major  ->  out: [2048, 512] f16 row-major (transposed)
__global__ __launch_bounds__(256)
void k_transpose_f16(const float* __restrict__ in, _Float16* __restrict__ out) {
    int i = blockIdx.x * 256 + threadIdx.x;          // over 2048*512
    if (i < G4H_ * H_) {
        int n = i >> 9;          // 0..2047
        int k = i & 511;         // 0..511
        out[i] = (_Float16)in[k * G4H_ + n];
    }
}

// ---- phase 1: G[s,b,n] = sum_k Xh[m=b*S+s, k] * WT[n, k] ------------------
// One wave computes a 16(M) x 64(N) strip: 4 accumulators, 16 K-steps.

__global__ __launch_bounds__(256)
void k_gemm(const _Float16* __restrict__ A,   // [M_, H_]   f16 (X rows, m = b*S+s)
            const _Float16* __restrict__ BT,  // [G4H_, H_] f16 (weight transposed)
            float* __restrict__ G)            // [S_, B_, G4H_] fp32
{
    const int wave  = (blockIdx.x << 3) + (threadIdx.x >> 5);   // 0..65535
    const int lane  = threadIdx.x & 31;
    const int tileM = wave >> 5;                                 // 0..2047
    const int strip = wave & 31;                                 // 0..31
    const int lr    = lane & 15;
    const int khalf = lane >> 4;

    const _Float16* Arow = A + (size_t)(tileM * 16 + lr) * H_;

    v8f acc[4];
    #pragma unroll
    for (int j = 0; j < 4; ++j)
        #pragma unroll
        for (int r = 0; r < 8; ++r) acc[j][r] = 0.0f;

    #pragma unroll 4
    for (int kb = 0; kb < 16; ++kb) {
        const int kbase = kb * 32;
        // A fragment (contiguous K: lanes 0-15 K=[0..15], lanes 16-31 K=[16..31])
        v16h a = *(const v16h*)(Arow + kbase + khalf * 16);
        __builtin_prefetch(Arow + kbase + 64, 0, 1);   // global_prefetch next K slab

        const _Float16* Bbase =
            BT + (size_t)(strip * 64 + lr) * H_ + kbase + khalf * 16;
        #pragma unroll
        for (int j = 0; j < 4; ++j) {
            v16h bfrag = *(const v16h*)(Bbase + (size_t)j * 16 * H_);
            acc[j] = __builtin_amdgcn_wmma_f32_16x16x32_f16(
                false, a, false, bfrag, (short)0, acc[j], false, false);
        }
    }

    // Scatter-store D tiles into G laid out [S, B, 4H]: m = b*S + s.
    #pragma unroll
    for (int j = 0; j < 4; ++j) {
        const int col = strip * 64 + j * 16 + lr;
        #pragma unroll
        for (int r = 0; r < 8; ++r) {
            const int m = tileM * 16 + khalf * 8 + r;
            const int s = m & (S_ - 1);
            const int b = m >> 10;
            G[(size_t)(s * B_ + b) * G4H_ + col] = acc[j][r];
        }
    }
}

// ---- phase 2: persistent recurrence ---------------------------------------
// 32 WGs x 256 threads. WG wg owns h columns [wg*16, wg*16+16).
// Wave v: mh = v&1 selects b-rows {0..15 | 16..31}; gate = v>>1 in {i,f,g,o}.

__global__ __launch_bounds__(256)
void k_rec(const float* __restrict__ G,       // [S_, B_, G4H_]
           const _Float16* __restrict__ UT,   // [G4H_, H_] f16 transposed
           _Float16* __restrict__ hbuf,       // [2][B_][H_] f16 double buffer
           unsigned* __restrict__ barCnt,
           _Float16* __restrict__ Ynext,      // [B_,S_,H_] f16 (next-layer input) or unused
           float* __restrict__ Yout,          // [B_,S_,H_] fp32 (final output) or unused
           float* __restrict__ Hout,          // [B_,H_] or unused
           float* __restrict__ Cout,          // [B_,H_] or unused
           int finalLayer)
{
    __shared__ float gatesL[4][B_][16];   // 8 KB
    __shared__ float cbufL[B_][16];       // 2 KB, persistent cell state slice

    const int tid   = threadIdx.x;
    const int wg    = blockIdx.x;         // 0..31
    const int wave  = tid >> 5;           // 0..7
    const int lane  = tid & 31;
    const int mh    = wave & 1;
    const int gate  = wave >> 1;          // 0..3
    const int lr    = lane & 15;
    const int khalf = lane >> 4;
    const int brow0 = mh * 16;

    // h_0 = 0 (buffer 0, own column slice), c_0 = 0
    for (int e = tid; e < B_ * 16; e += 256) {
        const int b = e >> 4, c = e & 15;
        hbuf[b * H_ + wg * 16 + c] = (_Float16)0.0f;
        cbufL[b][c] = 0.0f;
    }
    unsigned barIdx = 0;
    gridbar(barCnt, ++barIdx);

    const int n = gate * H_ + wg * 16 + lr;           // column in [0,2048)
    const _Float16* Brow = UT + (size_t)n * H_;

    for (int t = 0; t < S_; ++t) {
        const _Float16* hcur = hbuf + (size_t)(t & 1) * (B_ * H_);
        _Float16*       hnxt = hbuf + (size_t)((t + 1) & 1) * (B_ * H_);

        // Seed accumulator with precomputed X@Wx tile (free C operand).
        const float* Gt = G + (size_t)t * B_ * G4H_;
        v8f acc;
        #pragma unroll
        for (int r = 0; r < 8; ++r)
            acc[r] = Gt[(size_t)(brow0 + khalf * 8 + r) * G4H_ + n];

        // acc += h_t @ U   (K = 512 = 16 WMMA k-steps)
        const _Float16* Arow = hcur + (size_t)(brow0 + lr) * H_;
        #pragma unroll
        for (int kb = 0; kb < 16; ++kb) {
            const int kbase = kb * 32;
            v16h a = *(const v16h*)(Arow + kbase + khalf * 16);
            v16h bfrag = *(const v16h*)(Brow + kbase + khalf * 16);
            acc = __builtin_amdgcn_wmma_f32_16x16x32_f16(
                false, a, false, bfrag, (short)0, acc, false, false);
        }

        // Publish gate tile to LDS.
        #pragma unroll
        for (int r = 0; r < 8; ++r)
            gatesL[gate][brow0 + khalf * 8 + r][lr] = acc[r];
        __syncthreads();

        // LSTM cell elementwise for this WG's 32x16 slice.
        for (int e = tid; e < B_ * 16; e += 256) {
            const int b = e >> 4, c = e & 15;
            const float iv = sigmoidf_(gatesL[0][b][c]);
            const float fv = sigmoidf_(gatesL[1][b][c]);
            const float gv = tanhf(gatesL[2][b][c]);
            const float ov = sigmoidf_(gatesL[3][b][c]);
            const float cv = fv * cbufL[b][c] + iv * gv;
            const float hv = ov * tanhf(cv);
            cbufL[b][c] = cv;
            const int hcol = wg * 16 + c;
            hnxt[b * H_ + hcol] = (_Float16)hv;
            const size_t yidx = ((size_t)b * S_ + t) * H_ + hcol;
            if (finalLayer) {
                Yout[yidx] = hv;
                if (t == S_ - 1) {
                    Hout[b * H_ + hcol] = hv;
                    Cout[b * H_ + hcol] = cv;
                }
            } else {
                Ynext[yidx] = (_Float16)hv;
            }
        }
        // One barrier per step: makes h_{t+1} (other buffer) globally visible
        // and protects gatesL reuse via its internal __syncthreads.
        gridbar(barCnt, ++barIdx);
    }
}

// ---------------------------------------------------------------------------

extern "C" void kernel_launch(void* const* d_in, const int* in_sizes, int n_in,
                              void* d_out, int out_size, void* d_ws, size_t ws_size,
                              hipStream_t stream) {
    const float* x  = (const float*)d_in[0];   // [B,S,H]
    const float* W0 = (const float*)d_in[1];   // [H,4H]
    const float* W  = (const float*)d_in[2];   // [H,4H]
    const float* U  = (const float*)d_in[3];   // [H,4H]
    // d_in[4] = num_layers (== 2 per reference setup; hardcoded)

    float* out  = (float*)d_out;
    float* Yout = out;                                   // [B,S,H]
    float* Hout = out + (size_t)B_ * S_ * H_;            // [B,H]
    float* Cout = Hout + (size_t)B_ * H_;                // [B,H]

    // Workspace carve-out (256B aligned blocks).
    char* ws = (char*)d_ws;
    size_t off = 0;
    auto carve = [&](size_t bytes) -> void* {
        void* p = ws + off;
        off = (off + bytes + 255) & ~(size_t)255;
        return p;
    };
    unsigned*  cnt  = (unsigned*)carve(256);                          // 2 counters
    _Float16*  hbuf = (_Float16*)carve((size_t)2 * B_ * H_ * 2);      // 64 KB
    _Float16*  Xh0  = (_Float16*)carve((size_t)M_ * H_ * 2);          // 32 MB
    _Float16*  Xh1  = (_Float16*)carve((size_t)M_ * H_ * 2);          // 32 MB
    _Float16*  W0T  = (_Float16*)carve((size_t)G4H_ * H_ * 2);        // 2 MB
    _Float16*  WT   = (_Float16*)carve((size_t)G4H_ * H_ * 2);        // 2 MB
    _Float16*  UT   = (_Float16*)carve((size_t)G4H_ * H_ * 2);        // 2 MB
    float*     G    = (float*)carve((size_t)S_ * B_ * G4H_ * 4);      // 256 MB

    const int nX = M_ * H_;            // 16,777,216
    const int nW = G4H_ * H_;          // 1,048,576

    k_zero_counters<<<1, 64, 0, stream>>>(cnt);
    k_f32_to_f16<<<(nX + 255) / 256, 256, 0, stream>>>(x, Xh0, nX);
    k_transpose_f16<<<(nW + 255) / 256, 256, 0, stream>>>(W0, W0T);
    k_transpose_f16<<<(nW + 255) / 256, 256, 0, stream>>>(W,  WT);
    k_transpose_f16<<<(nW + 255) / 256, 256, 0, stream>>>(U,  UT);

    // Layer 0
    k_gemm<<<(M_ / 16) * (G4H_ / 64) / 8, 256, 0, stream>>>(Xh0, W0T, G);
    k_rec<<<NWG_REC, 256, 0, stream>>>(G, UT, hbuf, cnt + 0,
                                       Xh1, Yout, Hout, Cout, 0);
    // Layer 1 (final)
    k_gemm<<<(M_ / 16) * (G4H_ / 64) / 8, 256, 0, stream>>>(Xh1, WT, G);
    k_rec<<<NWG_REC, 256, 0, stream>>>(G, UT, hbuf, cnt + 1,
                                       Xh1, Yout, Hout, Cout, 1);
}
